// AttentionDecoder_20993800142968
// MI455X (gfx1250) — compile-verified, matrix operand order checked by reading, not run on hardware
//
#include <hip/hip_runtime.h>

// ---------------------------------------------------------------------------
// AttentionDecoder on MI455X (gfx1250, wave32, WMMA)
//
// B=64, S=512, H=512, T=64.  All GEMMs are X[M,K] @ W[N,K]^T form.
// bf16 WMMA 16x16x32 with f32 accumulation; f32 recurrent state/softmax.
// Working set (~150MB) is L2-resident on MI455X (192MB L2).
// Attention tanh sweep (1.07G transcendentals total) uses v_exp_f32-based
// tanh so TRANS ops co-execute with VALU; score phase runs on a Bx8 grid.
// ---------------------------------------------------------------------------

#define BB 64
#define SS 512
#define HH 512
#define TT 64

typedef __attribute__((ext_vector_type(16))) __bf16 v16bf;
typedef __attribute__((ext_vector_type(8)))  float  v8f;

union FragBF16 {
    unsigned int u[8];
    v16bf        v;
};

__device__ __forceinline__ unsigned short f32_to_bf16(float f) {
    unsigned int u = __float_as_uint(f);
    // round-to-nearest-even
    u += 0x7FFFu + ((u >> 16) & 1u);
    return (unsigned short)(u >> 16);
}

// tanh(x) = 1 - 2/(exp(2x)+1): one v_exp_f32 + one v_rcp_f32.
// Saturates correctly: x->+inf => 1, x->-inf => -1.
__device__ __forceinline__ float fast_tanh(float x) {
    float e = __expf(2.0f * x);
    return 1.0f - 2.0f * __builtin_amdgcn_rcpf(e + 1.0f);
}

__device__ __forceinline__ float fast_sigmoid(float x) {
    return __builtin_amdgcn_rcpf(1.0f + __expf(-x));
}

// ---------------------------------------------------------------------------
// Generic GEMM:  C[M,N] = X[M,K](bf16, ldx) @ W[N,K](bf16, ldw)^T    (f32 out)
// Block = 256 threads = 8 waves; wave grid 4(M) x 2(N) -> 64x32 block tile.
// Each wave computes a 16x16 tile with v_wmma_f32_16x16x32_bf16, K-chunks of 32.
// A-fragment layout (16-bit A 16x32): lane l holds row (l&15);
//   VGPR v holds K pair k = (v<4 ? 2v : 8+2v) + ((l>=16) ? 8 : 0).
// B operand of W^T uses the identical loader over W's rows (N as "M").
// ---------------------------------------------------------------------------
__global__ void gemm_xwt_bf16(const unsigned short* __restrict__ X, int ldx,
                              const unsigned short* __restrict__ W, int ldw,
                              float* __restrict__ C, int ldc, int K) {
    const int lane = threadIdx.x & 31;
    const int wave = threadIdx.x >> 5;
    const int m0 = blockIdx.x * 64 + (wave & 3) * 16;
    const int n0 = blockIdx.y * 32 + (wave >> 2) * 16;

    const int r     = lane & 15;
    const int half8 = (lane >> 4) << 3;   // 0 or 8

    int kof[8];
#pragma unroll
    for (int v = 0; v < 8; ++v)
        kof[v] = ((v < 4) ? (2 * v) : (8 + 2 * v)) + half8;

    const unsigned short* xrow = X + (size_t)(m0 + r) * (size_t)ldx;
    const unsigned short* wrow = W + (size_t)(n0 + r) * (size_t)ldw;

    v8f acc = {};
    for (int k = 0; k < K; k += 32) {
        FragBF16 a, b;
#pragma unroll
        for (int v = 0; v < 8; ++v) {
            a.u[v] = *(const unsigned int*)(xrow + k + kof[v]);
            b.u[v] = *(const unsigned int*)(wrow + k + kof[v]);
        }
        acc = __builtin_amdgcn_wmma_f32_16x16x32_bf16(
            false, a.v, false, b.v, (short)0, acc, false, false);
    }

    // C/D layout: VGPR i -> M = i + 8*(lane>=16), N = lane&15
    const int n    = n0 + (lane & 15);
    const int mrow = m0 + ((lane >> 4) << 3);
#pragma unroll
    for (int i = 0; i < 8; ++i)
        C[(size_t)(mrow + i) * (size_t)ldc + n] = acc[i];
}

// ---------------------------------------------------------------------------
// Setup kernels
// ---------------------------------------------------------------------------
__global__ void cast_f32_bf16(const float* __restrict__ src,
                              unsigned short* __restrict__ dst, long n) {
    long stride = (long)gridDim.x * blockDim.x;
    for (long i = (long)blockIdx.x * blockDim.x + threadIdx.x; i < n; i += stride)
        dst[i] = f32_to_bf16(src[i]);
}

// dst[rows, Ka+Kb] = [A(rows,Ka) | Bm(rows,Kb)] cast to bf16
__global__ void concat_w_bf16(const float* __restrict__ A, int Ka,
                              const float* __restrict__ Bm, int Kb,
                              unsigned short* __restrict__ dst, long total) {
    const int Kt = Ka + Kb;
    long stride = (long)gridDim.x * blockDim.x;
    for (long i = (long)blockIdx.x * blockDim.x + threadIdx.x; i < total; i += stride) {
        long row = i / Kt;
        int  col = (int)(i - row * Kt);
        float v = (col < Ka) ? A[row * Ka + col] : Bm[row * Kb + (col - Ka)];
        dst[i] = f32_to_bf16(v);
    }
}

// xcat0[b,1536] = [dec_in=0 | context | h0]; xcat1[b,1024] = [h0n | h1]
__global__ void init_state(const float* __restrict__ h00, const float* __restrict__ c00,
                           const float* __restrict__ h01, const float* __restrict__ c01,
                           unsigned short* __restrict__ xcat0,
                           unsigned short* __restrict__ xcat1,
                           float* __restrict__ c0, float* __restrict__ c1) {
    int i = blockIdx.x * blockDim.x + threadIdx.x;
    if (i >= BB * HH) return;
    int b = i / HH, h = i - b * HH;
    xcat0[(size_t)b * 1536 + h]        = 0;                      // dec_in = 0
    xcat0[(size_t)b * 1536 + 512 + h]  = 0;                      // context slot
    xcat0[(size_t)b * 1536 + 1024 + h] = f32_to_bf16(h00[i]);    // h0
    xcat1[(size_t)b * 1024 + h]        = 0;                      // h0n slot
    xcat1[(size_t)b * 1024 + 512 + h]  = f32_to_bf16(h01[i]);    // h1
    c0[i] = c00[i];
    c1[i] = c00 == c01 ? c01[i] : c01[i]; // keep simple; both read once
    c1[i] = c01[i];
}

// ---------------------------------------------------------------------------
// Attention phase 1: scores.  Grid (B, 8); each block covers 64 s-values,
// one wave per s (8 waves).  Loop-invariant (hproj+ba) and v are cached in
// 16 registers per lane, so the inner loop streams only enc_proj.
//   score[b,s] = sum_h tanh(enc_proj[b,s,h] + hproj[b,h] + ba[h]) * v[h]
// ---------------------------------------------------------------------------
__global__ void attn_scores(const float* __restrict__ enc_proj,
                            const float* __restrict__ hproj,
                            const float* __restrict__ ba,
                            const float* __restrict__ vvec,
                            float* __restrict__ scores) {
    const int b     = blockIdx.x;
    const int chunk = blockIdx.y;          // 0..7 -> 64 s each
    const int lane  = threadIdx.x & 31;
    const int wave  = threadIdx.x >> 5;

    const float* hp = hproj + (size_t)b * HH;
    float hb[16], vv[16];
#pragma unroll
    for (int j = 0; j < 16; ++j) {
        const int h = lane + 32 * j;
        hb[j] = hp[h] + ba[h];
        vv[j] = vvec[h];
    }

    const int s0 = chunk * 64;
    for (int s = s0 + wave; s < s0 + 64; s += 8) {
        const float* row = enc_proj + ((size_t)b * SS + s) * HH;
        float acc = 0.0f;
#pragma unroll
        for (int j = 0; j < 16; ++j)
            acc += fast_tanh(row[lane + 32 * j] + hb[j]) * vv[j];
#pragma unroll
        for (int off = 16; off; off >>= 1)
            acc += __shfl_xor(acc, off, 32);
        if (lane == 0) scores[(size_t)b * SS + s] = acc;
    }
}

// ---------------------------------------------------------------------------
// Attention phase 2: softmax over S + context.  One block per b.
//   w = softmax(scores[b,:]);  context[h] = sum_s w[s]*enc[b,s,h]
// Writes context (bf16) into the x_cat0 context slot.
// ---------------------------------------------------------------------------
__global__ void attn_softmax_ctx(const float* __restrict__ scores,
                                 const float* __restrict__ enc,
                                 unsigned short* __restrict__ ctx_dst, int ctx_ld) {
    __shared__ float sc[SS];
    __shared__ float red[8];

    const int b    = blockIdx.x;
    const int tid  = threadIdx.x;
    const int lane = tid & 31;
    const int wave = tid >> 5;

    for (int s = tid; s < SS; s += 256) sc[s] = scores[(size_t)b * SS + s];
    __syncthreads();

    // max
    float m = -3.402823466e+38f;
    for (int s = tid; s < SS; s += 256) m = fmaxf(m, sc[s]);
#pragma unroll
    for (int off = 16; off; off >>= 1) m = fmaxf(m, __shfl_xor(m, off, 32));
    if (lane == 0) red[wave] = m;
    __syncthreads();
    m = red[0];
#pragma unroll
    for (int i = 1; i < 8; ++i) m = fmaxf(m, red[i]);
    __syncthreads();

    // exp + sum
    float sum = 0.0f;
    for (int s = tid; s < SS; s += 256) {
        float e = __expf(sc[s] - m);
        sc[s] = e;
        sum += e;
    }
#pragma unroll
    for (int off = 16; off; off >>= 1) sum += __shfl_xor(sum, off, 32);
    if (lane == 0) red[wave] = sum;
    __syncthreads();
    float total = 0.0f;
#pragma unroll
    for (int i = 0; i < 8; ++i) total += red[i];
    const float rinv = __builtin_amdgcn_rcpf(total);
    __syncthreads();

    // context: threads sweep h (coalesced), loop s (sc broadcast from LDS)
    const float* er = enc + (size_t)b * SS * HH;
    for (int h = tid; h < HH; h += 256) {
        float acc = 0.0f;
        for (int s = 0; s < SS; ++s)
            acc += sc[s] * er[(size_t)s * HH + h];
        ctx_dst[(size_t)b * ctx_ld + h] = f32_to_bf16(acc * rinv);
    }
}

// ---------------------------------------------------------------------------
// LSTM pointwise, layer 0: gates[b,4H] (pre-bias), torch gate order i,f,g,o.
// Writes h_new (bf16) into two destination slots; updates c in place (f32).
// ---------------------------------------------------------------------------
__global__ void lstm0_pointwise(const float* __restrict__ gates,
                                const float* __restrict__ bias,
                                float* __restrict__ c,
                                unsigned short* __restrict__ hdst0, int ld0,
                                unsigned short* __restrict__ hdst1, int ld1) {
    const int b = blockIdx.x;
    const float* g = gates + (size_t)b * 4 * HH;
    for (int h = threadIdx.x; h < HH; h += 256) {
        float ig = fast_sigmoid(g[h]          + bias[h]);
        float fg = fast_sigmoid(g[HH + h]     + bias[HH + h]);
        float gg = fast_tanh   (g[2 * HH + h] + bias[2 * HH + h]);
        float og = fast_sigmoid(g[3 * HH + h] + bias[3 * HH + h]);
        float cn = fg * c[(size_t)b * HH + h] + ig * gg;
        c[(size_t)b * HH + h] = cn;
        unsigned short hb = f32_to_bf16(og * fast_tanh(cn));
        hdst0[(size_t)b * ld0 + h] = hb;
        hdst1[(size_t)b * ld1 + h] = hb;
    }
}

// Layer 1: same, plus y[b] = dot(h_new, Wf) + bf -> out[b*T + t]
__global__ void lstm1_pointwise(const float* __restrict__ gates,
                                const float* __restrict__ bias,
                                float* __restrict__ c,
                                unsigned short* __restrict__ hdst0, int ld0,
                                unsigned short* __restrict__ hdst1, int ld1,
                                const float* __restrict__ Wf,
                                const float* __restrict__ bf,
                                float* __restrict__ out, int t) {
    __shared__ float red[8];
    const int b    = blockIdx.x;
    const int lane = threadIdx.x & 31;
    const int wave = threadIdx.x >> 5;
    const float* g = gates + (size_t)b * 4 * HH;
    float acc = 0.0f;
    for (int h = threadIdx.x; h < HH; h += 256) {
        float ig = fast_sigmoid(g[h]          + bias[h]);
        float fg = fast_sigmoid(g[HH + h]     + bias[HH + h]);
        float gg = fast_tanh   (g[2 * HH + h] + bias[2 * HH + h]);
        float og = fast_sigmoid(g[3 * HH + h] + bias[3 * HH + h]);
        float cn = fg * c[(size_t)b * HH + h] + ig * gg;
        c[(size_t)b * HH + h] = cn;
        float hn = og * fast_tanh(cn);
        unsigned short hb = f32_to_bf16(hn);
        hdst0[(size_t)b * ld0 + h] = hb;
        hdst1[(size_t)b * ld1 + h] = hb;
        acc += hn * Wf[h];
    }
#pragma unroll
    for (int off = 16; off; off >>= 1) acc += __shfl_xor(acc, off, 32);
    if (lane == 0) red[wave] = acc;
    __syncthreads();
    if (threadIdx.x == 0) {
        float tot = 0.0f;
#pragma unroll
        for (int i = 0; i < 8; ++i) tot += red[i];
        out[(size_t)b * TT + t] = tot + bf[0];
    }
}

// ---------------------------------------------------------------------------
// Host-side orchestration
// ---------------------------------------------------------------------------
static inline size_t align256(size_t x) { return (x + 255) & ~(size_t)255; }

extern "C" void kernel_launch(void* const* d_in, const int* in_sizes, int n_in,
                              void* d_out, int out_size, void* d_ws, size_t ws_size,
                              hipStream_t stream) {
    (void)in_sizes; (void)n_in; (void)out_size; (void)ws_size;

    const float* enc   = (const float*)d_in[0];   // [B,S,H]
    const float* h0_0  = (const float*)d_in[1];
    const float* c0_0  = (const float*)d_in[2];
    const float* h0_1  = (const float*)d_in[3];
    const float* c0_1  = (const float*)d_in[4];
    const float* Wa_h  = (const float*)d_in[5];   // [H,H]
    const float* Wa_e  = (const float*)d_in[6];   // [H,H]
    const float* ba    = (const float*)d_in[7];   // [H]
    const float* vvec  = (const float*)d_in[8];   // [H]
    const float* Wih0  = (const float*)d_in[9];   // [4H,2H]
    const float* Whh0  = (const float*)d_in[10];  // [4H,H]
    const float* b_0   = (const float*)d_in[11];  // [4H]
    const float* Wih1  = (const float*)d_in[12];  // [4H,H]
    const float* Whh1  = (const float*)d_in[13];  // [4H,H]
    const float* b_1   = (const float*)d_in[14];  // [4H]
    const float* Wf    = (const float*)d_in[15];  // [1,H]
    const float* bfb   = (const float*)d_in[16];  // [1]
    float* out = (float*)d_out;                   // [B,T,1]

    // workspace carve-up
    char* p = (char*)d_ws;
    size_t off = 0;
    unsigned short* enc_bf = (unsigned short*)(p + off); off = align256(off + (size_t)BB * SS * HH * 2);
    float*          eproj  = (float*)(p + off);          off = align256(off + (size_t)BB * SS * HH * 4);
    unsigned short* Wah_bf = (unsigned short*)(p + off); off = align256(off + (size_t)HH * HH * 2);
    unsigned short* Wae_bf = (unsigned short*)(p + off); off = align256(off + (size_t)HH * HH * 2);
    unsigned short* Wcat0  = (unsigned short*)(p + off); off = align256(off + (size_t)4 * HH * 3 * HH * 2); // [2048,1536]
    unsigned short* Wcat1  = (unsigned short*)(p + off); off = align256(off + (size_t)4 * HH * 2 * HH * 2); // [2048,1024]
    unsigned short* xcat0  = (unsigned short*)(p + off); off = align256(off + (size_t)BB * 3 * HH * 2);     // [64,1536]
    unsigned short* xcat1  = (unsigned short*)(p + off); off = align256(off + (size_t)BB * 2 * HH * 2);     // [64,1024]
    float*          gates0 = (float*)(p + off);          off = align256(off + (size_t)BB * 4 * HH * 4);
    float*          gates1 = (float*)(p + off);          off = align256(off + (size_t)BB * 4 * HH * 4);
    float*          hproj  = (float*)(p + off);          off = align256(off + (size_t)BB * HH * 4);
    float*          score  = (float*)(p + off);          off = align256(off + (size_t)BB * SS * 4);
    float*          c0     = (float*)(p + off);          off = align256(off + (size_t)BB * HH * 4);
    float*          c1     = (float*)(p + off);          off = align256(off + (size_t)BB * HH * 4);

    // ---- setup: casts, weight concat, state init ----
    cast_f32_bf16<<<4096, 256, 0, stream>>>(enc, enc_bf, (long)BB * SS * HH);
    cast_f32_bf16<<<512, 256, 0, stream>>>(Wa_h, Wah_bf, (long)HH * HH);
    cast_f32_bf16<<<512, 256, 0, stream>>>(Wa_e, Wae_bf, (long)HH * HH);
    concat_w_bf16<<<2048, 256, 0, stream>>>(Wih0, 2 * HH, Whh0, HH, Wcat0, (long)4 * HH * 3 * HH);
    concat_w_bf16<<<2048, 256, 0, stream>>>(Wih1, HH, Whh1, HH, Wcat1, (long)4 * HH * 2 * HH);
    init_state<<<(BB * HH + 255) / 256, 256, 0, stream>>>(h0_0, c0_0, h0_1, c0_1,
                                                          xcat0, xcat1, c0, c1);

    // ---- one-time: enc_proj = enc @ Wa_e^T  (32768 x 512 x 512, WMMA) ----
    {
        dim3 grid(BB * SS / 64, HH / 32);
        gemm_xwt_bf16<<<grid, 256, 0, stream>>>(enc_bf, HH, Wae_bf, HH, eproj, HH, HH);
    }

    // ---- decode loop ----
    for (int t = 0; t < TT; ++t) {
        // hproj = h1 @ Wa_h^T   (h1 lives in xcat1 slot 1, row stride 1024)
        {
            dim3 grid(1, HH / 32);
            gemm_xwt_bf16<<<grid, 256, 0, stream>>>(xcat1 + HH, 2 * HH,
                                                    Wah_bf, HH, hproj, HH, HH);
        }
        // attention scores (B x 8 grid), then softmax+context per b
        {
            dim3 grid(BB, 8);
            attn_scores<<<grid, 256, 0, stream>>>(eproj, hproj, ba, vvec, score);
        }
        attn_softmax_ctx<<<BB, 256, 0, stream>>>(score, enc, xcat0 + HH, 3 * HH);
        // gates0 = xcat0[64,1536] @ Wcat0[2048,1536]^T
        {
            dim3 grid(1, 4 * HH / 32);
            gemm_xwt_bf16<<<grid, 256, 0, stream>>>(xcat0, 3 * HH,
                                                    Wcat0, 3 * HH, gates0, 4 * HH, 3 * HH);
        }
        // LSTM0 pointwise: h0n -> xcat0 slot 2 (next-step recurrent), xcat1 slot 0
        lstm0_pointwise<<<BB, 256, 0, stream>>>(gates0, b_0, c0,
                                                xcat0 + 2 * HH, 3 * HH,
                                                xcat1, 2 * HH);
        // gates1 = xcat1[64,1024] @ Wcat1[2048,1024]^T
        {
            dim3 grid(1, 4 * HH / 32);
            gemm_xwt_bf16<<<grid, 256, 0, stream>>>(xcat1, 2 * HH,
                                                    Wcat1, 2 * HH, gates1, 4 * HH, 2 * HH);
        }
        // LSTM1 pointwise: h1n -> xcat1 slot 1 (recurrent+attn), xcat0 slot 0 (dec_in);
        // y[b,t] = h1n . Wf + bf
        lstm1_pointwise<<<BB, 256, 0, stream>>>(gates1, b_1, c1,
                                                xcat1 + HH, 2 * HH,
                                                xcat0, 3 * HH,
                                                Wf, bfb, out, t);
    }
}
